// SelectiveSSM_24567212933754
// MI455X (gfx1250) — compile-verified, matrix operand order checked by reading, not run on hardware
//
#include <hip/hip_runtime.h>
#include <hip/hip_bf16.h>
#include <math.h>

// ---------------------------------------------------------------------------
// Selective SSM (Mamba-style) block for MI455X / gfx1250, wave32 + WMMA bf16.
// B=4, L=1024, D=1024, N=16.
//  - weights pre-packed to bf16 in WMMA fragment order (one-shot kernel)
//  - GEMM: single-barrier software pipeline; A and B tiles double-buffered in
//    LDS; B streamed with global_load_async_to_lds_b128 (ASYNCcnt)
// ---------------------------------------------------------------------------

#define DM    1024          // d_model
#define NS    16            // d_state
#define BB    4             // batch
#define LL    1024          // seq len
#define MT    (BB*LL)       // 4096 rows

typedef __bf16 bf16;
typedef __attribute__((ext_vector_type(16))) __bf16 v16bf;
typedef __attribute__((ext_vector_type(4)))  __bf16 v4bf;
typedef __attribute__((ext_vector_type(8)))  float  v8f;

__device__ __forceinline__ bf16 f2bf(float f) {
    unsigned u = __builtin_bit_cast(unsigned, f);
    u += 0x7fffu + ((u >> 16) & 1u);           // round-to-nearest-even
    unsigned short h = (unsigned short)(u >> 16);
    return __builtin_bit_cast(bf16, h);
}

// WMMA 16-bit A/B fragment addressing (CDNA5 ISA 7.12.2):
// lane = (idx&15) + 16*kh with kh = bit3 of k; half index j: k=(j<8?j:j+8)+kh*8.
__device__ __forceinline__ int frag_j(int k, int& kh) {
    kh = (k >> 3) & 1;
    int t = k - 8 * kh;
    return (t < 8) ? t : (t - 8);
}

// Async copy: 16 contiguous global bytes -> 16 contiguous LDS bytes, per lane.
__device__ __forceinline__ void async_ld_b128(unsigned lds_off, const void* g) {
    asm volatile("global_load_async_to_lds_b128 %0, %1, off"
                 :: "v"(lds_off), "v"(g) : "memory");
}

enum { EPI_NONE = 0, EPI_SOFTPLUS = 1, EPI_ADD = 2 };

// ---------------------------------------------------------------------------
// One-shot weight pack: W[K x N] f32 row-major -> bf16 blocks, 8KB per
// (n-tile, k-tile), laid out in exact WMMA B-fragment order so the GEMM can
// async-memcpy a tile straight into LDS. Block id = nt*ktiles + kt.
// ---------------------------------------------------------------------------
__global__ __launch_bounds__(256)
void pack_weight(const float* __restrict__ W, int ld, int N, int ktiles,
                 bf16* __restrict__ P)
{
    const int blk = blockIdx.x;
    const int kt  = blk % ktiles;
    const int nt  = blk / ktiles;
    const int t   = threadIdx.x;
    const int nf   = t >> 5;
    const int lane = t & 31;
    const int kh   = lane >> 4;
    const int n    = nt * 128 + nf * 16 + (lane & 15);

    v16bf frag;
    #pragma unroll
    for (int j = 0; j < 16; ++j) {
        int k = kt * 32 + ((j < 8) ? j : j + 8) + kh * 8;
        float v = (n < N) ? W[(size_t)k * ld + n] : 0.f;
        frag[j] = f2bf(v);
    }
    *((v16bf*)(P + (size_t)blk * 4096) + t) = frag;
}

// ---------------------------------------------------------------------------
// C[M x Ntot] = epi( A[M x K] (f32, stride lda, col off acol) @ W )
// Single-barrier pipeline per K-step (32 k), 128x128 tile / 256 threads:
//   issue A(kt+1) global loads + B(kt+1) async copy -> other buffer
//   s_wait_asynccnt 2 (retire B(kt)) ; __syncthreads()
//   8 WMMAs on buffers[kt&1]
//   convert + ds_store A(kt+1) into buffers[(kt+1)&1]   (no barrier needed)
// ---------------------------------------------------------------------------
template<int EPI>
__global__ __launch_bounds__(256)
void gemm_bf16_wmma(const float* __restrict__ Ag, int lda, int acol,
                    const bf16* __restrict__ Wp,        // packed [nt][kt][4096]
                    const float* __restrict__ bias,
                    const float* __restrict__ aux, int ldaux,
                    float* __restrict__ Cg, int ldc,
                    int K, int Ntot)
{
    __shared__ alignas(32) bf16 Abuf[2][4096];     // 2 x 8 KB
    __shared__ alignas(32) bf16 Bbuf[2][4096];     // 2 x 8 KB

    const int tid    = threadIdx.x;
    const int wave   = tid >> 5;
    const int lane   = tid & 31;
    const int m_base = blockIdx.y * 128;
    const int n_base = blockIdx.x * 128;
    const int ktiles = K >> 5;

    const bf16* wt = Wp + (size_t)blockIdx.x * ktiles * 4096;
    const unsigned ldsB0 = (unsigned)(size_t)(void*)&Bbuf[0][0];
    const unsigned ldsB1 = (unsigned)(size_t)(void*)&Bbuf[1][0];

    // per-thread A staging geometry: linear = i*1024 + tid*4
    int a_row[4], a_k[4], a_dst[4];
    #pragma unroll
    for (int i = 0; i < 4; ++i) {
        int linear = i * 1024 + tid * 4;
        a_row[i] = linear >> 5;
        a_k[i]   = linear & 31;                    // multiple of 4
        int kh; int j = frag_j(a_k[i], kh);
        a_dst[i] = ((a_row[i] >> 4) * 32 + (a_row[i] & 15) + 16 * kh) * 16 + j;
    }
    const float* arow_base = Ag + (size_t)m_base * lda + acol;

    v8f acc[8] = {};
    float4 areg[4];

    // ---- prologue: A0 -> regs -> Abuf[0];  B0 async -> Bbuf[0] ----
    #pragma unroll
    for (int i = 0; i < 4; ++i)
        areg[i] = *(const float4*)(arow_base + (size_t)a_row[i] * lda + a_k[i]);
    {
        const char* g = (const char*)wt + tid * 32;
        async_ld_b128(ldsB0 + tid * 32,      g);
        async_ld_b128(ldsB0 + tid * 32 + 16, g + 16);
    }
    #pragma unroll
    for (int i = 0; i < 4; ++i) {
        v4bf pkv;
        pkv[0] = f2bf(areg[i].x); pkv[1] = f2bf(areg[i].y);
        pkv[2] = f2bf(areg[i].z); pkv[3] = f2bf(areg[i].w);
        *(v4bf*)&Abuf[0][a_dst[i]] = pkv;
    }

    for (int kt = 0; kt < ktiles; ++kt) {
        const int cur = kt & 1;
        const bool more = (kt + 1 < ktiles);

        if (more) {
            // next A tile -> registers (latency hidden behind WMMAs)
            const float* ab = arow_base + (kt + 1) * 32;
            #pragma unroll
            for (int i = 0; i < 4; ++i)
                areg[i] = *(const float4*)(ab + (size_t)a_row[i] * lda + a_k[i]);
            // next B tile -> other LDS buffer, async
            const char* g = (const char*)(wt + (size_t)(kt + 1) * 4096) + tid * 32;
            unsigned l = (cur ? ldsB0 : ldsB1) + tid * 32;
            async_ld_b128(l,      g);
            async_ld_b128(l + 16, g + 16);
            asm volatile("s_wait_asynccnt 2" ::: "memory");  // retire B(kt)
        } else {
            asm volatile("s_wait_asynccnt 0" ::: "memory");
        }
        __syncthreads();   // A(kt)/B(kt) visible to all waves

        // ---- 8 WMMAs per wave per K-step ----
        const bf16* bsel = &Bbuf[cur][0];
        v16bf af = *((const v16bf*)&Abuf[cur][0] + wave * 32 + lane);
        #pragma unroll
        for (int nf = 0; nf < 8; ++nf) {
            v16bf bfm = *((const v16bf*)bsel + nf * 32 + lane);
            acc[nf] = __builtin_amdgcn_wmma_f32_16x16x32_bf16(
                          false, af, false, bfm, (short)0, acc[nf], false, false);
        }

        if (more) {
            // convert + store next A tile into the other buffer; the next
            // iteration's barrier orders these before any wave reads them.
            #pragma unroll
            for (int i = 0; i < 4; ++i) {
                v4bf pkv;
                pkv[0] = f2bf(areg[i].x); pkv[1] = f2bf(areg[i].y);
                pkv[2] = f2bf(areg[i].z); pkv[3] = f2bf(areg[i].w);
                *(v4bf*)&Abuf[cur ^ 1][a_dst[i]] = pkv;
            }
        }
    }

    // ---- epilogue + store (C/D layout: VGPR r, lane l -> M = r + 8*(l>>4), N = l&15) ----
    const int half = lane >> 4;
    const int nn   = lane & 15;
    #pragma unroll
    for (int nf = 0; nf < 8; ++nf) {
        int col = n_base + nf * 16 + nn;
        if (col >= Ntot) continue;
        #pragma unroll
        for (int r = 0; r < 8; ++r) {
            int row = m_base + wave * 16 + half * 8 + r;
            float c = acc[nf][r];
            if (EPI == EPI_SOFTPLUS) {
                c += bias[col];
                c = (c > 20.f) ? c : log1pf(__expf(c));
            } else if (EPI == EPI_ADD) {
                c += aux[(size_t)row * ldaux + col];
            }
            Cg[(size_t)row * ldc + col] = c;
        }
    }
}

// ---------------------------------------------------------------------------
// Selective scan, fused with skip (x*D) and silu(z) gate. One (b,d) channel
// per 16-lane half-wave: lane n holds state h[n]; shfl-xor reduction for y_t.
// ---------------------------------------------------------------------------
__global__ __launch_bounds__(128)
void scan_fused(const float* __restrict__ xz,     // [MT, 2048]
                const float* __restrict__ proj,   // [MT, 1056]
                const float* __restrict__ delta,  // [MT, 1024]
                const float* __restrict__ A_log,  // [1024, 16]
                const float* __restrict__ Dp,     // [1024]
                float* __restrict__ u)            // [MT, 1024]
{
    const int tid = threadIdx.x;
    const int grp = tid >> 4;                  // 8 channels per block
    const int n   = tid & 15;
    const int ch  = blockIdx.x * 8 + grp;      // 0..4095
    const int b   = ch >> 10;
    const int d   = ch & 1023;

    const float An   = -__expf(A_log[d * NS + n]);
    const float dpar = Dp[d];

    float h = 0.f;
    const size_t rowbase = (size_t)b * LL;
    for (int t = 0; t < LL; ++t) {
        size_t r  = rowbase + t;
        float dt  = delta[r * 1024 + d];
        float x   = xz[r * 2048 + d];
        float z   = xz[r * 2048 + 1024 + d];
        float Bn  = proj[r * 1056 + 1024 + n];
        float Cn  = proj[r * 1056 + 1040 + n];
        float dA  = __expf(dt * An);
        h = dA * h + (dt * x) * Bn;
        float c = h * Cn;
        #pragma unroll
        for (int off = 8; off > 0; off >>= 1) c += __shfl_xor(c, off, 16);
        if (n == 0) {
            float y   = c + x * dpar;
            float sig = 1.f / (1.f + __expf(-z));
            u[r * 1024 + d] = y * (z * sig);
        }
    }
}

// ---------------------------------------------------------------------------
// In-place row LayerNorm over D=1024, one block per row.
// ---------------------------------------------------------------------------
__global__ __launch_bounds__(256)
void layernorm_inplace(float* __restrict__ io,
                       const float* __restrict__ g,
                       const float* __restrict__ bta)
{
    __shared__ float s_sum[8], s_sq[8];
    const int row = blockIdx.x;
    const int tid = threadIdx.x;
    float* p = io + (size_t)row * DM;

    float4 v = *((const float4*)p + tid);
    float s = v.x + v.y + v.z + v.w;
    float q = v.x * v.x + v.y * v.y + v.z * v.z + v.w * v.w;
    #pragma unroll
    for (int off = 16; off > 0; off >>= 1) {
        s += __shfl_xor(s, off, 32);
        q += __shfl_xor(q, off, 32);
    }
    if ((tid & 31) == 0) { s_sum[tid >> 5] = s; s_sq[tid >> 5] = q; }
    __syncthreads();
    if (tid < 8) {
        s = s_sum[tid]; q = s_sq[tid];
        #pragma unroll
        for (int off = 4; off > 0; off >>= 1) {
            s += __shfl_xor(s, off, 8);
            q += __shfl_xor(q, off, 8);
        }
        if (tid == 0) { s_sum[0] = s; s_sq[0] = q; }
    }
    __syncthreads();
    const float mu   = s_sum[0] * (1.f / DM);
    const float var  = s_sq[0] * (1.f / DM) - mu * mu;
    const float rstd = rsqrtf(var + 1e-5f);

    float4 gv = *((const float4*)g + tid);
    float4 bv = *((const float4*)bta + tid);
    float4 o;
    o.x = (v.x - mu) * rstd * gv.x + bv.x;
    o.y = (v.y - mu) * rstd * gv.y + bv.y;
    o.z = (v.z - mu) * rstd * gv.z + bv.z;
    o.w = (v.w - mu) * rstd * gv.w + bv.w;
    *((float4*)p + tid) = o;
}

extern "C" void kernel_launch(void* const* d_in, const int* in_sizes, int n_in,
                              void* d_out, int out_size, void* d_ws, size_t ws_size,
                              hipStream_t stream)
{
    const float* x_in   = (const float*)d_in[0];   // [4,1024,1024]
    const float* W_in   = (const float*)d_in[1];   // [1024,2048]
    const float* W_x    = (const float*)d_in[2];   // [1024,1056]
    const float* W_dt   = (const float*)d_in[3];   // [1024,1024]
    const float* b_dt   = (const float*)d_in[4];   // [1024]
    const float* A_log  = (const float*)d_in[5];   // [1024,16]
    const float* D_par  = (const float*)d_in[6];   // [1024]
    const float* W_out  = (const float*)d_in[7];   // [1024,1024]
    const float* ln_g   = (const float*)d_in[8];   // [1024]
    const float* ln_b   = (const float*)d_in[9];   // [1024]
    float* out = (float*)d_out;                    // [4,1024,1024]

    // ---- workspace layout ----
    float* ws    = (float*)d_ws;
    float* xz    = ws;                             // [4096, 2048]
    float* proj  = xz    + (size_t)MT * 2048;      // [4096, 1056]
    float* delta = proj  + (size_t)MT * 1056;      // [4096, 1024]
    float* u     = delta + (size_t)MT * 1024;      // [4096, 1024]
    bf16* pk     = (bf16*)(u + (size_t)MT * 1024); // packed weights (bf16)
    bf16* Win_p  = pk;                             // 16*32 blocks
    bf16* Wx_p   = Win_p + (size_t)16 * 32 * 4096; //  9*32 blocks
    bf16* Wdt_p  = Wx_p  + (size_t) 9 * 32 * 4096; //  8*32 blocks
    bf16* Wout_p = Wdt_p + (size_t) 8 * 32 * 4096; //  8*32 blocks

    const dim3 blk(256);

    // 0) pack all weights to bf16 WMMA-fragment order (one pass each)
    pack_weight<<<dim3(16 * 32), blk, 0, stream>>>(W_in,  2048, 2048, 32, Win_p);
    pack_weight<<<dim3( 9 * 32), blk, 0, stream>>>(W_x,   1056, 1056, 32, Wx_p);
    pack_weight<<<dim3( 8 * 32), blk, 0, stream>>>(W_dt,  1024, 1024, 32, Wdt_p);
    pack_weight<<<dim3( 8 * 32), blk, 0, stream>>>(W_out, 1024, 1024, 32, Wout_p);

    const dim3 gXZ (2048 / 128, MT / 128);
    const dim3 gPRJ((1056 + 127) / 128, MT / 128);
    const dim3 gSQ (1024 / 128, MT / 128);

    // 1) xz = x_in @ W_in
    gemm_bf16_wmma<EPI_NONE><<<gXZ, blk, 0, stream>>>(
        x_in, 1024, 0, Win_p, nullptr, nullptr, 0, xz, 2048, 1024, 2048);

    // 2) proj = x @ W_x   (x = xz[:, :1024])
    gemm_bf16_wmma<EPI_NONE><<<gPRJ, blk, 0, stream>>>(
        xz, 2048, 0, Wx_p, nullptr, nullptr, 0, proj, 1056, 1024, 1056);

    // 3) delta = softplus(delta_raw @ W_dt + b_dt)   (delta_raw = proj[:, :1024])
    gemm_bf16_wmma<EPI_SOFTPLUS><<<gSQ, blk, 0, stream>>>(
        proj, 1056, 0, Wdt_p, b_dt, nullptr, 0, delta, 1024, 1024, 1024);

    // 4) selective scan + skip + silu gate -> u
    scan_fused<<<dim3(512), dim3(128), 0, stream>>>(xz, proj, delta, A_log, D_par, u);

    // 5) res = u @ W_out + x_in  -> d_out
    gemm_bf16_wmma<EPI_ADD><<<gSQ, blk, 0, stream>>>(
        u, 1024, 0, Wout_p, nullptr, x_in, 1024, out, 1024, 1024, 1024);

    // 6) layernorm in-place on d_out
    layernorm_inplace<<<dim3(MT), dim3(256), 0, stream>>>(out, ln_g, ln_b);
}
